// SFVoxelModel_88785563943602
// MI455X (gfx1250) — matrix operand order repaired and checked by reading, not compiled, not run on hardware
//
#include <hip/hip_runtime.h>
#include <hip/hip_fp16.h>

// Problem constants from setup_inputs(): B=4, C=64, H=W=512, V=8192, M=8, N=64
#define BDIM 4
#define CDIM 64
#define HDIM 512
#define WDIM 512
#define VDIM 8192
#define MDIM 8
#define NDIM 64
#define EPSF 1e-8f

typedef __attribute__((ext_vector_type(16))) _Float16 v16h;
typedef __attribute__((ext_vector_type(8)))  _Float16 v8h;
typedef __attribute__((ext_vector_type(8)))  float    v8f;

// ---------------------------------------------------------------------------
// Kernel 1: gather voxel features from the pseudo-image, quantize to f16 for
// the WMMA phase, and store the RECIPROCAL of the clamped L2 norm (f32).
// One wave32 per (array, voxel); lane l handles channels l and l+32.
// Waves [0, B*V) -> src array, waves [B*V, 2*B*V) -> dst array.
// ---------------------------------------------------------------------------
__global__ __launch_bounds__(256) void sfv_extract_kernel(
    const float* __restrict__ image,      // [B, C, H, W]
    const int*   __restrict__ vox_src,    // [B, V, 2]
    const int*   __restrict__ vox_dst,    // [B, V, 2]
    _Float16*    __restrict__ fsrc,       // [B*V, C] out
    _Float16*    __restrict__ fdst,       // [B*V, C] out
    float*       __restrict__ inv_na,     // [B*V]    out (1 / clamped norm)
    float*       __restrict__ inv_nb)     // [B*V]    out (1 / clamped norm)
{
    const int wave  = (blockIdx.x * 256 + threadIdx.x) >> 5;
    const int lane  = threadIdx.x & 31;
    const int total = BDIM * VDIM;
    const bool isDst = wave >= total;
    const int id = isDst ? (wave - total) : wave;

    const int* __restrict__ vox = isDst ? vox_dst : vox_src;
    _Float16*  __restrict__ fo  = isDst ? fdst : fsrc;
    float*     __restrict__ no  = isDst ? inv_nb : inv_na;

    const int r = vox[2 * id];
    const int c = vox[2 * id + 1];
    const bool valid = (r >= 0) && (c >= 0);
    const int b = id / VDIM;
    const size_t pix   = (size_t)r * WDIM + (size_t)c;
    const size_t ibase = (size_t)b * CDIM * (size_t)(HDIM * WDIM);

    float v0 = -1.0f, v1 = -1.0f;            // reference fills -1.0 when invalid
    if (valid) {
        v0 = image[ibase + (size_t)lane        * (HDIM * WDIM) + pix];
        v1 = image[ibase + (size_t)(lane + 32) * (HDIM * WDIM) + pix];
    }

    // Wave32 sum of squares over all 64 channels.
    float ss = v0 * v0 + v1 * v1;
#pragma unroll
    for (int off = 16; off >= 1; off >>= 1)
        ss += __shfl_xor(ss, off, 32);

    fo[(size_t)id * CDIM + lane]      = (_Float16)v0;
    fo[(size_t)id * CDIM + lane + 32] = (_Float16)v1;
    if (lane == 0)
        no[id] = 1.0f / fmaxf(sqrtf(ss), EPSF);
}

// ---------------------------------------------------------------------------
// Kernel 2: per src voxel, corr[v, n] = <s_v, g_{knn_dst[v,n]}> * inva * invb.
// One wave32 per src voxel; 64x64 matvec on the matrix pipe:
//   A (16x32 f16) = 16 gathered dst-neighbor rows (one K-chunk of 32)
//   B (32x16 f16) = src vector chunk replicated into all 16 columns
//   D (16x16 f32) accumulated over 2 K-chunks -> 8 x v_wmma_f32_16x16x32_f16.
// VGPR layouts per CDNA5 ISA 7.12.2 (A: lane half selects K sub-ranges;
// this code feeds A and B alike in k = 16*half + s order; D: group0 rows 0-7, group1 rows 8-15).
// Masking is folded into the per-neighbor reciprocal norm (invalid -> 0), so
// the WMMA->store path is branch-free and EXEC stays all-ones around WMMA.
// ---------------------------------------------------------------------------
__global__ __launch_bounds__(256) void sfv_corr_wmma_kernel(
    const _Float16* __restrict__ fsrc,     // [B*V, C]
    const _Float16* __restrict__ fdst,     // [B*V, C]
    const float*    __restrict__ inv_na,   // [B*V]
    const float*    __restrict__ inv_nb,   // [B*V]
    const int*      __restrict__ knn_dst,  // [B*V, N]
    float*          __restrict__ corr)     // [B*V, N] out
{
    const int bv    = (blockIdx.x * 256 + threadIdx.x) >> 5;  // grid == B*V waves
    const int lane  = threadIdx.x & 31;
    const int half  = lane >> 4;        // 0: lanes 0-15, 1: lanes 16-31
    const int mrow  = lane & 15;        // A-matrix row this lane owns
    const int r7    = lane & 7;
    const int bbase = (bv / VDIM) * VDIM;

    // B fragments (src vector, columns replicated): contiguous 16 halves.
    const _Float16* s = fsrc + (size_t)bv * CDIM;
    const v16h bf0 = *(const v16h*)(s + half * 16);        // K chunk 0: K=0..31
    const v16h bf1 = *(const v16h*)(s + 32 + half * 16);   // K chunk 1: K=32..63

    const float  inva   = inv_na[bv];
    const size_t knbase = (size_t)bv * NDIM;

#pragma unroll
    for (int t = 0; t < 4; ++t) {                 // 4 M-tiles of 16 neighbors
        const int nidx = knn_dst[knbase + t * 16 + mrow];
        // Reciprocal dst norm for this lane's neighbor; 0 kills masked outputs.
        float invb = 0.0f;
        if (nidx >= 0) invb = inv_nb[bbase + nidx];

        // A rows: no need to zero invalid rows — D rows are independent and a
        // zero reciprocal annihilates that row's output at the write below.
        const _Float16* g =
            fdst + (size_t)(bbase + (nidx < 0 ? 0 : nidx)) * CDIM;
        const v16h a0 = *(const v16h*)(g + half * 16);        // K = h*16 .. +15
        const v16h a1 = *(const v16h*)(g + 32 + half * 16);   // K = 32+h*16 .. +15

        v8f c = {};
        c = __builtin_amdgcn_wmma_f32_16x16x32_f16(
                false, a0, false, bf0, (short)0, c, false, false);
        c = __builtin_amdgcn_wmma_f32_16x16x32_f16(
                false, a1, false, bf1, (short)0, c, false, false);

        // Extract row value (half*8 + r7): unrolled select (v_cndmask chain).
        float x = c[0];
        x = (r7 == 1) ? c[1] : x;
        x = (r7 == 2) ? c[2] : x;
        x = (r7 == 3) ? c[3] : x;
        x = (r7 == 4) ? c[4] : x;
        x = (r7 == 5) ? c[5] : x;
        x = (r7 == 6) ? c[6] : x;
        x = (r7 == 7) ? c[7] : x;

        // This lane writes row rr = half*8 + r7; its invb lives in lane rr.
        const float invbr = __shfl(invb, half * 8 + r7, 32);
        const float val   = x * inva * invbr;

        if (mrow < 8)   // 16 writer lanes -> one coalesced 64B store per tile
            corr[knbase + t * 16 + half * 8 + r7] = val;
    }
}

// ---------------------------------------------------------------------------
// Kernel 3: inflate corr through knn_idxs_src -> out [B, V, M, N].
// float4-vectorized: one thread per 4 consecutive n values; fully coalesced
// b128 reads/writes for the 64 MB output stream.
// ---------------------------------------------------------------------------
__global__ __launch_bounds__(256) void sfv_inflate_kernel(
    const float4* __restrict__ corr4,     // [B*V, N/4]
    const int*    __restrict__ knn_src,   // [B*V, M]
    float4*       __restrict__ out4)      // [B*V, M, N/4]
{
    const size_t o4 = (size_t)blockIdx.x * 256 + threadIdx.x;
    const int    n4 = (int)(o4 & (NDIM / 4 - 1));   // 16 float4 per row
    const size_t t  = o4 >> 4;                      // = bv*M + m
    const int    m  = (int)(t & (MDIM - 1));
    const size_t bv = t >> 3;
    const int bbase = (int)((bv / VDIM) * VDIM);

    const int idx = knn_src[bv * MDIM + m];
    float4 val = make_float4(0.0f, 0.0f, 0.0f, 0.0f);
    if (idx >= 0)
        val = corr4[(size_t)(bbase + idx) * (NDIM / 4) + n4];
    out4[o4] = val;
}

// ---------------------------------------------------------------------------
extern "C" void kernel_launch(void* const* d_in, const int* in_sizes, int n_in,
                              void* d_out, int out_size, void* d_ws, size_t ws_size,
                              hipStream_t stream) {
    const float* image   = (const float*)d_in[0];
    const int*   vox_src = (const int*)d_in[1];
    const int*   vox_dst = (const int*)d_in[2];
    const int*   knn_src = (const int*)d_in[3];
    const int*   knn_dst = (const int*)d_in[4];
    float* out = (float*)d_out;

    // Workspace layout (all offsets 4KiB+ aligned):
    //   fsrc   f16 [B*V*C]  : 4 MiB  @ 0
    //   fdst   f16 [B*V*C]  : 4 MiB  @ 4 MiB
    //   inv_na f32 [B*V]    : 128KiB @ 8 MiB
    //   inv_nb f32 [B*V]    : 128KiB @ 8 MiB + 128 KiB
    //   corr   f32 [B*V*N]  : 8 MiB  @ 8 MiB + 256 KiB
    char* ws = (char*)d_ws;
    _Float16* fsrc   = (_Float16*)(ws);
    _Float16* fdst   = (_Float16*)(ws + (size_t)(4u << 20));
    float*    inv_na = (float*)   (ws + (size_t)(8u << 20));
    float*    inv_nb = (float*)   (ws + (size_t)(8u << 20) + (128u << 10));
    float*    corr   = (float*)   (ws + (size_t)(8u << 20) + (256u << 10));

    // Kernel 1: 2*B*V waves = 65536 -> 8192 blocks of 8 waves.
    sfv_extract_kernel<<<(2 * BDIM * VDIM) / 8, 256, 0, stream>>>(
        image, vox_src, vox_dst, fsrc, fdst, inv_na, inv_nb);

    // Kernel 2: B*V waves = 32768 -> 4096 blocks (exact; EXEC all-ones).
    sfv_corr_wmma_kernel<<<(BDIM * VDIM) / 8, 256, 0, stream>>>(
        fsrc, fdst, inv_na, inv_nb, knn_dst, corr);

    // Kernel 3: one thread per float4 of output (B*V*M*N/4 = 4194304).
    sfv_inflate_kernel<<<(BDIM * VDIM * MDIM * (NDIM / 4)) / 256, 256, 0, stream>>>(
        (const float4*)corr, knn_src, (float4*)out);
}
